// Router_34832184770693
// MI455X (gfx1250) — matrix operand order repaired and checked by reading, not compiled, not run on hardware
//
#include <hip/hip_runtime.h>
#include <hip/hip_bf16.h>

#define D_MODEL   2816
#define N_EXPERTS 64
#define TOPK      8
#define RMS_EPS   1e-6f
#define KCHUNKS   (D_MODEL / 32)     // 88
#define NTILES    (N_EXPERTS / 16)   // 4
#define INV_SQRT_D 0.018844442f      // 2816^-0.5

typedef __attribute__((ext_vector_type(16))) _Float16 v16h;
typedef __attribute__((ext_vector_type(8)))  float    v8f;
typedef __attribute__((ext_vector_type(4)))  float    v4f;

// ---------------------------------------------------------------------------
// Prepack: Wp[kc][nt][lane][j] = (W[n][k] * scale[k]) as f16, in the exact
// B-fragment layout this code uses for 16-bit B (32x16 K-major tile, contiguous k as in A):
//   lane l -> column n = nt*16 + (l&15),  K = kc*32 + (l>>4)*16 + j,  j=0..15
// One lane's 16 halfs are contiguous (32B) -> main kernel loads are trivial.
// ---------------------------------------------------------------------------
__global__ void router_prepack(const float* __restrict__ W,
                               const float* __restrict__ scale,
                               _Float16* __restrict__ Wp) {
  int tid = blockIdx.x * blockDim.x + threadIdx.x;
  if (tid >= KCHUNKS * NTILES * 32) return;
  int lane = tid & 31;
  int nt   = (tid >> 5) & 3;
  int kc   = tid >> 7;
  int n     = nt * 16 + (lane & 15);
  int kbase = kc * 32 + (lane >> 4) * 16;
  const float* wrow = W + (size_t)n * D_MODEL + kbase;
  const float* srow = scale + kbase;
  _Float16* dst = Wp + (size_t)tid * 16;
#pragma unroll
  for (int j = 0; j < 16; ++j)
    dst[j] = (_Float16)(wrow[j] * srow[j]);
}

// ---------------------------------------------------------------------------
// Main: fused RMSNorm + expert-logit GEMM (WMMA f16->f32) + softmax/top-8 +
// renorm * per_expert_scale + dense scatter. One pass over x (HBM-bound).
// 8 waves/block, each wave: 16 tokens x 64 experts.
// ---------------------------------------------------------------------------
__global__ __launch_bounds__(256) void router_main(
    const float* __restrict__ x, const _Float16* __restrict__ Wp,
    const float* __restrict__ pes, float* __restrict__ out) {
  __shared__ float sc[8][16][65];  // +1 pad: avoid bank conflicts

  const int lane  = threadIdx.x & 31;
  const int wave  = threadIdx.x >> 5;
  const int row   = lane & 15;     // token within tile / expert within tile
  const int khalf = lane >> 4;     // which K half-pattern this lane holds
  const int mBase = (blockIdx.x * 8 + wave) * 16;
  const float* xrow = x + (size_t)(mBase + row) * D_MODEL;

  v8f acc0 = {}, acc1 = {}, acc2 = {}, acc3 = {};
  float sumsq = 0.f;

  for (int kc = 0; kc < D_MODEL; kc += 32) {
    __builtin_prefetch(xrow + kc + 512, 0, 1);  // stream-ahead on x
    // A fragment (16-bit A 16x32 layout): lane-half khalf holds
    // K = kc + khalf*16 + {0..15} (contiguous, matches prepacked B)
    const float* pa = xrow + kc + khalf * 16;
    __attribute__((aligned(16))) float af[16];
    *(v4f*)(af + 0)  = *(const v4f*)(pa + 0);
    *(v4f*)(af + 4)  = *(const v4f*)(pa + 4);
    *(v4f*)(af + 8)  = *(const v4f*)(pa + 8);
    *(v4f*)(af + 12) = *(const v4f*)(pa + 12);
    v16h a;
#pragma unroll
    for (int j = 0; j < 16; ++j) {
      float v = af[j];
      sumsq = __builtin_fmaf(v, v, sumsq);  // fused RMS accumulation
      a[j] = (_Float16)v;
    }
    // B fragments: contiguous 32B per lane from the prepacked layout
    const v16h* bp = (const v16h*)Wp + (size_t)(kc >> 5) * (NTILES * 32) + lane;
    v16h b0 = bp[0 * 32];
    v16h b1 = bp[1 * 32];
    v16h b2 = bp[2 * 32];
    v16h b3 = bp[3 * 32];
    acc0 = __builtin_amdgcn_wmma_f32_16x16x32_f16(false, a, false, b0, (short)0, acc0, false, false);
    acc1 = __builtin_amdgcn_wmma_f32_16x16x32_f16(false, a, false, b1, (short)0, acc1, false, false);
    acc2 = __builtin_amdgcn_wmma_f32_16x16x32_f16(false, a, false, b2, (short)0, acc2, false, false);
    acc3 = __builtin_amdgcn_wmma_f32_16x16x32_f16(false, a, false, b3, (short)0, acc3, false, false);
  }

  // Finish per-token sum of squares: lanes l and l+16 hold complementary K.
  sumsq += __shfl_xor(sumsq, 16, 32);
  // Combined RMSNorm * d^-0.5 factor for token (lane%16)
  float fsc = rsqrtf(sumsq * (1.0f / D_MODEL) + RMS_EPS) * INV_SQRT_D;

  // Scatter C tiles to LDS with per-row scaling.
  // C layout: VGPR r -> (M = r + 8*khalf, N = lane%16).
#pragma unroll
  for (int r = 0; r < 8; ++r) {
    int m = r + khalf * 8;
    float fm = __shfl(fsc, m, 32);
    sc[wave][m][ 0 + row] = acc0[r] * fm;
    sc[wave][m][16 + row] = acc1[r] * fm;
    sc[wave][m][32 + row] = acc2[r] * fm;
    sc[wave][m][48 + row] = acc3[r] * fm;
  }
  __syncthreads();  // cross-lane LDS handoff

  // Softmax + top-8 + renorm + per-expert scale. softmax denominator cancels
  // against top-k renormalization: w_i = exp(s_i-max) / sum_top8 exp(s_j-max).
  if (lane < 16) {
    float* s = &sc[wave][lane][0];
    float mx = s[0];
#pragma unroll
    for (int e = 1; e < N_EXPERTS; ++e) mx = fmaxf(mx, s[e]);
    int idx[TOPK]; float wv[TOPK]; float wsum = 0.f;
#pragma unroll
    for (int k = 0; k < TOPK; ++k) {
      float best = -3.402823e38f; int bi = 0;
#pragma unroll
      for (int e = 0; e < N_EXPERTS; ++e) {
        float v = s[e];
        if (v > best) { best = v; bi = e; }   // strict '>': first-index ties
      }
      s[bi] = -3.402823e38f;
      float ev = __expf(best - mx);
      idx[k] = bi; wv[k] = ev; wsum += ev;
    }
    float inv = 1.0f / wsum;
#pragma unroll
    for (int e = 0; e < N_EXPERTS; ++e) s[e] = 0.f;
#pragma unroll
    for (int k = 0; k < TOPK; ++k) s[idx[k]] = wv[k] * inv * pes[idx[k]];
  }
  __syncthreads();

  // Coalesced dense store: 16 tokens x 64 experts per wave.
  float* ob = out + (size_t)mBase * N_EXPERTS;
  for (int i = lane; i < 16 * N_EXPERTS; i += 32) {
    int m = i >> 6, e = i & 63;
    ob[i] = sc[wave][m][e];
  }
}

extern "C" void kernel_launch(void* const* d_in, const int* in_sizes, int n_in,
                              void* d_out, int out_size, void* d_ws, size_t ws_size,
                              hipStream_t stream) {
  const float* x     = (const float*)d_in[0];
  const float* W     = (const float*)d_in[1];
  const float* scale = (const float*)d_in[2];
  const float* pes   = (const float*)d_in[3];
  _Float16* Wp = (_Float16*)d_ws;  // 64*2816*2 = 352 KB scratch

  int pk_threads = KCHUNKS * NTILES * 32;  // 11264
  router_prepack<<<(pk_threads + 255) / 256, 256, 0, stream>>>(W, scale, Wp);

  int tokens = in_sizes[0] / D_MODEL;      // 32768
  router_main<<<tokens / 128, 256, 0, stream>>>(x, Wp, pes, (float*)d_out);
}
